// MultiHeadAttention_9483287790076
// MI455X (gfx1250) — matrix operand order repaired and checked by reading, not compiled, not run on hardware
//
#include <hip/hip_runtime.h>
#include <hip/hip_bf16.h>

typedef __attribute__((ext_vector_type(16))) _Float16 v16h;
typedef __attribute__((ext_vector_type(8)))  _Float16 v8h;
typedef __attribute__((ext_vector_type(4)))  _Float16 v4h;
typedef __attribute__((ext_vector_type(8)))  float    v8f;

#define CONCAT16(lo, hi) __builtin_shufflevector((lo), (hi), 0,1,2,3,4,5,6,7,8,9,10,11,12,13,14,15)

constexpr int S_LEN = 1024, NH = 12, DK = 64, DM = 768;

// ---------------------------------------------------------------------------
// Generic tiled GEMM:  out[M,N] = A[M,K] (f32 or f16) * W[K,N] (f32) + bias[N]
// MODE 0: f16 row-major out.  MODE 1: f16 out transposed per-head [b][h][d][s].
// MODE 2: f32 row-major out.
// Block: 128 threads (4 waves). Tile 64x64, BK=64. wave w -> rows w*16..w*16+15.
// ---------------------------------------------------------------------------
template <typename TIn, int MODE>
__global__ void __launch_bounds__(128)
gemm_kernel(const TIn* __restrict__ A, const float* __restrict__ W,
            const float* __restrict__ bias, void* __restrict__ outp) {
    constexpr int N = 768, K = 768;
    __shared__ __align__(16) _Float16 As[64][72];   // [m][k], padded (144B rows)
    __shared__ __align__(16) _Float16 Bs[64][72];   // [n][k] (transposed), padded

    const int tid  = threadIdx.x;
    const int wave = tid >> 5;
    const int lane = tid & 31;
    const int ln   = lane & 15;
    const int hi   = lane >> 4;           // 0: lanes 0-15, 1: lanes 16-31
    const int m0   = blockIdx.y * 64;
    const int n0   = blockIdx.x * 64;
    const int ab   = hi * 8;              // unused here (A-frag also uses kl, 0/16)
    const int kl   = hi * 16;             // B-frag K base within 32-chunk (0/16)

    v8f c[4] = {};                        // 16x16 tiles along N

    for (int k0 = 0; k0 < K; k0 += 64) {
        __syncthreads();
        // ---- stage A tile (64x64) into LDS as f16 ----
        if constexpr (sizeof(TIn) == 4) {
            #pragma unroll
            for (int i = 0; i < 8; ++i) {
                const int e = tid * 32 + i * 4;        // 4096 elems / 128 thr
                const int row = e >> 6, col = e & 63;
                const float4 f = *(const float4*)((const float*)A +
                                   (size_t)(m0 + row) * K + k0 + col);
                v4h hv = {(_Float16)f.x, (_Float16)f.y, (_Float16)f.z, (_Float16)f.w};
                *(v4h*)&As[row][col] = hv;
            }
        } else {
            #pragma unroll
            for (int i = 0; i < 4; ++i) {
                const int e = tid * 32 + i * 8;
                const int row = e >> 6, col = e & 63;
                *(v8h*)&As[row][col] = *(const v8h*)((const _Float16*)A +
                                        (size_t)(m0 + row) * K + k0 + col);
            }
        }
        // ---- stage W tile (64x64) into LDS transposed [n][k] as f16 ----
        #pragma unroll
        for (int i = 0; i < 8; ++i) {
            const int e = tid * 32 + i * 4;
            const int kr = e >> 6, nc = e & 63;
            const float4 f = *(const float4*)(W + (size_t)(k0 + kr) * N + n0 + nc);
            Bs[nc + 0][kr] = (_Float16)f.x;
            Bs[nc + 1][kr] = (_Float16)f.y;
            Bs[nc + 2][kr] = (_Float16)f.z;
            Bs[nc + 3][kr] = (_Float16)f.w;
        }
        __syncthreads();

        // ---- fragments + WMMA (8 per wave per barrier interval) ----
        const int m = wave * 16 + ln;
        #pragma unroll
        for (int cch = 0; cch < 2; ++cch) {
            const v8h alo = *(const v8h*)&As[m][cch * 32 + kl];
            const v8h ahi = *(const v8h*)&As[m][cch * 32 + kl + 8];
            const v16h af = CONCAT16(alo, ahi);
            #pragma unroll
            for (int nt = 0; nt < 4; ++nt) {
                const v16h bf = *(const v16h*)&Bs[nt * 16 + ln][cch * 32 + kl];
                c[nt] = __builtin_amdgcn_wmma_f32_16x16x32_f16(
                            false, af, false, bf, (short)0, c[nt], false, false);
            }
        }
    }

    // ---- epilogue: bias + store ----
    #pragma unroll
    for (int nt = 0; nt < 4; ++nt) {
        const int n = n0 + nt * 16 + ln;
        const float bv = bias[n];
        #pragma unroll
        for (int v = 0; v < 8; ++v) {
            const int mg = m0 + wave * 16 + v + 8 * hi;   // C row = VGPR + 8*half
            const float val = c[nt][v] + bv;
            if constexpr (MODE == 0) {
                ((_Float16*)outp)[(size_t)mg * N + n] = (_Float16)val;
            } else if constexpr (MODE == 1) {
                const int bb = mg >> 10, s = mg & 1023;   // m = b*1024 + s
                const int hh = n >> 6,  d = n & 63;       // n = h*64 + d
                ((_Float16*)outp)[(((size_t)bb * NH + hh) * DK + d) * S_LEN + s] =
                    (_Float16)val;
            } else {
                ((float*)outp)[(size_t)mg * N + n] = val;
            }
        }
    }
}

// ---------------------------------------------------------------------------
// One 32-key block of flash attention for one wave (16 query rows).
// Scores computed TRANSPOSED (S^T = K @ Q^T): C-frag lane = query, rows = keys.
//  -> row max / row sum are in-lane + one xor-16 shuffle.
//  -> P A-fragment (queries x keys) is built fully in-register (no LDS).
// MASKED=false: keys fully visible (bulk phase), no causal compare/select.
// ---------------------------------------------------------------------------
template <bool MASKED>
__device__ __forceinline__ void attn_block(
    const int kb, const int qsel /* this lane's query */, const int ln,
    const int hi, const int ab, const int kl,
    const _Float16* __restrict__ Kbase, const _Float16* __restrict__ Vbase,
    const v16h (&qfb)[2], v8f (&acc)[4], float& mrow, float& lrow) {

    // ---- S^T tiles: keys kb..kb+15 (st0), kb+16..kb+31 (st1), cols = queries
    v8f st0 = {}, st1 = {};
    const _Float16* Kp0 = Kbase + (size_t)(kb + ln) * DM;        // A rows = keys
    const _Float16* Kp1 = Kbase + (size_t)(kb + 16 + ln) * DM;
    if constexpr (!MASKED) {                       // streaming phase: prefetch next
        __builtin_prefetch(Kp0 + 32 * DM, 0, 3);
        __builtin_prefetch(Kp1 + 32 * DM, 0, 3);
    }
    #pragma unroll
    for (int cch = 0; cch < 2; ++cch) {
        const v8h a0lo = *(const v8h*)(Kp0 + cch * 32 + ab);
        const v8h a0hi = *(const v8h*)(Kp0 + cch * 32 + ab + 16);
        const v8h a1lo = *(const v8h*)(Kp1 + cch * 32 + ab);
        const v8h a1hi = *(const v8h*)(Kp1 + cch * 32 + ab + 16);
        const v16h a0 = CONCAT16(a0lo, a0hi);
        const v16h a1 = CONCAT16(a1lo, a1hi);
        st0 = __builtin_amdgcn_wmma_f32_16x16x32_f16(
                  false, a0, false, qfb[cch], (short)0, st0, false, false);
        st1 = __builtin_amdgcn_wmma_f32_16x16x32_f16(
                  false, a1, false, qfb[cch], (short)0, st1, false, false);
    }

    // ---- scale + (mask): lane holds 16 keys of its own query ----
    float s0[8], s1[8];
    #pragma unroll
    for (int v = 0; v < 8; ++v) {
        s0[v] = st0[v] * 0.125f;                   // 1/sqrt(64)
        s1[v] = st1[v] * 0.125f;
        if constexpr (MASKED) {
            const int k0i = kb + v + 8 * hi;       // C row v -> key v + 8*half
            const int k1i = kb + 16 + v + 8 * hi;
            s0[v] = (k0i > qsel) ? -3.0e38f : s0[v];
            s1[v] = (k1i > qsel) ? -3.0e38f : s1[v];
        }
    }

    // ---- online softmax: in-lane reduce + single xor-16 combine ----
    float mt = fmaxf(s0[0], s1[0]);
    #pragma unroll
    for (int v = 1; v < 8; ++v) mt = fmaxf(mt, fmaxf(s0[v], s1[v]));
    mt = fmaxf(mt, __shfl_xor(mt, 16, 32));        // partner lane: other 16 keys
    const float mnew = fmaxf(mrow, mt);
    const float corr = __expf(mrow - mnew);
    mrow = mnew;

    float p0[8], p1[8], psum = 0.0f;
    #pragma unroll
    for (int v = 0; v < 8; ++v) {
        p0[v] = __expf(s0[v] - mnew);
        p1[v] = __expf(s1[v] - mnew);
        psum += p0[v] + p1[v];
    }
    psum += __shfl_xor(psum, 16, 32);
    lrow = lrow * corr + psum;

    // ---- P A-fragment, fully in-register: halves = keys {8hi..}, {8hi+16..}
    v16h pf;
    #pragma unroll
    for (int v = 0; v < 8; ++v) {
        pf[v]     = (_Float16)p0[v];               // keys kb + 8hi + v
        pf[v + 8] = (_Float16)p1[v];               // keys kb + 16 + 8hi + v
    }

    // ---- rescale O accumulator (rows = queries v+8hi: fetch their corr) ----
    #pragma unroll
    for (int v = 0; v < 8; ++v) {
        const float cr = __shfl(corr, v + 8 * hi, 32);
        #pragma unroll
        for (int dt = 0; dt < 4; ++dt) acc[dt][v] = acc[dt][v] * cr;
    }
    // ---- P @ V ----
    #pragma unroll
    for (int dt = 0; dt < 4; ++dt) {
        const _Float16* Vp = Vbase + (size_t)(dt * 16 + ln) * S_LEN + kb + ab;
        if constexpr (!MASKED) __builtin_prefetch(Vp + 32, 0, 3);
        const v16h vf = CONCAT16(*(const v8h*)Vp, *(const v8h*)(Vp + 16));
        acc[dt] = __builtin_amdgcn_wmma_f32_16x16x32_f16(
                      false, pf, false, vf, (short)0, acc[dt], false, false);
    }
}

// ---------------------------------------------------------------------------
// Flash attention: block = (q-tile of 64, head, batch), 4 waves x 16 q-rows.
// Q,K row-major f16 [b][s][h*64+d]; V transposed f16 [b][h][d][s]; O row-major.
// ---------------------------------------------------------------------------
__global__ void __launch_bounds__(128)
attn_kernel(const _Float16* __restrict__ Qg, const _Float16* __restrict__ Kg,
            const _Float16* __restrict__ Vt, _Float16* __restrict__ Og) {
    const int qblk = blockIdx.x, h = blockIdx.y, b = blockIdx.z;
    const int tid  = threadIdx.x;
    const int wave = tid >> 5;
    const int lane = tid & 31;
    const int ln   = lane & 15;
    const int hi   = lane >> 4;
    const int ab   = hi * 8;
    const int kl   = hi * 16;

    const int qw   = qblk * 64 + wave * 16;        // wave's first query row
    const int qsel = qw + ln;                      // this lane's query (S^T col)

    // ---- Q as B-fragments (Q^T): lane = query, halves = d {8hi..}, {8hi+16..} ----
    const _Float16* Qrow = Qg + ((size_t)(b * S_LEN + qsel)) * DM + h * DK;
    v16h qfb[2];
    #pragma unroll
    for (int cch = 0; cch < 2; ++cch)
        qfb[cch] = CONCAT16(*(const v8h*)(Qrow + cch * 32 + ab), *(const v8h*)(Qrow + cch * 32 + ab + 16));

    v8f acc[4] = {};
    float mrow = -3.0e38f, lrow = 0.0f;

    const _Float16* Kbase = Kg + ((size_t)b * S_LEN) * DM + h * DK;
    const _Float16* Vbase = Vt + ((size_t)((b * NH + h) * DK)) * S_LEN;
    const int kcausal = qblk * 64;                 // keys below this: no masking
    const int kend    = qblk * 64 + 64;

    int kb = 0;
    for (; kb < kcausal; kb += 32)                 // bulk: unmasked, prefetching
        attn_block<false>(kb, qsel, ln, hi, ab, kl, Kbase, Vbase,
                          qfb, acc, mrow, lrow);
    for (; kb < kend; kb += 32)                    // diagonal tail: causal mask
        attn_block<true>(kb, qsel, ln, hi, ab, kl, Kbase, Vbase,
                         qfb, acc, mrow, lrow);

    // ---- normalize + store (O rows = queries v+8hi) ----
    #pragma unroll
    for (int v = 0; v < 8; ++v) {
        const float lr  = __shfl(lrow, v + 8 * hi, 32);
        const float inv = 1.0f / lr;
        const int q = qw + v + 8 * hi;
        _Float16* Orow = Og + ((size_t)(b * S_LEN + q)) * DM + h * DK;
        #pragma unroll
        for (int dt = 0; dt < 4; ++dt)
            Orow[dt * 16 + ln] = (_Float16)(acc[dt][v] * inv);
    }
}

// ---------------------------------------------------------------------------
extern "C" void kernel_launch(void* const* d_in, const int* in_sizes, int n_in,
                              void* d_out, int out_size, void* d_ws, size_t ws_size,
                              hipStream_t stream) {
    (void)in_sizes; (void)n_in; (void)out_size; (void)ws_size;
    const float* x  = (const float*)d_in[0];
    // d_in[1] = mask (tril) — causality applied analytically in attn_kernel
    const float* Wq = (const float*)d_in[2];
    const float* bq = (const float*)d_in[3];
    const float* Wk = (const float*)d_in[4];
    const float* bk = (const float*)d_in[5];
    const float* Wv = (const float*)d_in[6];
    const float* bv = (const float*)d_in[7];
    const float* Wo = (const float*)d_in[8];
    const float* bo = (const float*)d_in[9];

    const size_t nElem = (size_t)8 * S_LEN * DM;        // per-tensor f16 elems
    _Float16* Qw  = (_Float16*)d_ws;                    // [b][s][h*64+d]
    _Float16* Kw  = Qw + nElem;                         // [b][s][h*64+d]
    _Float16* Vtw = Kw + nElem;                         // [b][h][d][s]
    _Float16* Ow  = Vtw + nElem;                        // [b][s][h*64+d]

    dim3 gg(DM / 64, (8 * S_LEN) / 64);                 // (N tiles, M tiles)
    gemm_kernel<float, 0><<<gg, 128, 0, stream>>>(x, Wq, bq, Qw);
    gemm_kernel<float, 0><<<gg, 128, 0, stream>>>(x, Wk, bk, Kw);
    gemm_kernel<float, 1><<<gg, 128, 0, stream>>>(x, Wv, bv, Vtw);

    dim3 ga(S_LEN / 64, NH, 8);                         // (q-tiles, heads, batch)
    attn_kernel<<<ga, 128, 0, stream>>>(Qw, Kw, Vtw, Ow);

    gemm_kernel<_Float16, 2><<<gg, 128, 0, stream>>>(Ow, Wo, bo, (float*)d_out);
}
